// MotivateHeroesCell_82695300317806
// MI455X (gfx1250) — compile-verified, matrix operand order checked by reading, not run on hardware
//
#include <hip/hip_runtime.h>
#include <hip/hip_bf16.h>

// ---------------------------------------------------------------------------
// MI455X (gfx1250) implementation. All big GEMMs use v_wmma_f32_16x16x32_bf16
// (f32 accumulate), staged through LDS with the CDNA5 wave32 WMMA lane layout.
// f32->bf16 conversion happens once at tile staging, stored as packed b32.
// Elementwise cell math stays f32.
// ---------------------------------------------------------------------------

typedef __attribute__((ext_vector_type(16))) __bf16 v16bf;
typedef __attribute__((ext_vector_type(8)))  float  v8f;

#define TM 128
#define TN 128
#define TK 32
#define LDSS 40   // TK + 8 pad (bf16 elements) -> 80B row stride, 16B aligned

static __device__ __forceinline__ unsigned short f2bf(float f) {
    unsigned u = __builtin_bit_cast(unsigned, f);
    u += 0x7FFFu + ((u >> 16) & 1u);          // round-to-nearest-even
    return (unsigned short)(u >> 16);
}

static __device__ __forceinline__ unsigned pack2bf(float a, float b) {
    return (unsigned)f2bf(a) | ((unsigned)f2bf(b) << 16);
}

union Frag16 { uint4 q[2]; v16bf v; };

// Accumulate a K-strip: acc += A[128 x K] * W[K x 128] (tile at m0,n0)
static __device__ __forceinline__ void gemm_ktiles(
    const float* __restrict__ A, int lda,
    const float* __restrict__ W, int ldw,
    int K, int m0, int n0,
    unsigned short* As, unsigned short* Bs,
    v8f acc[2][4])
{
    const int tid  = threadIdx.x;
    const int lane = tid & 31;
    const int wv   = tid >> 5;
    const int wm   = (wv & 3) * 32;   // wave row  (4 wave-rows x 32)
    const int wn   = (wv >> 2) * 64;  // wave col  (2 wave-cols x 64)
    const int lr   = lane & 15;
    const int kh   = (lane >> 4) * 8; // K-half per ISA 16-bit A/B layout

    // B-tile staging map: thread owns one column n, one 16-wide k-group
    const int bn = tid & 127;
    const int bk = (tid >> 7) * 16;

    for (int k0 = 0; k0 < K; k0 += TK) {
        // --- stage A tile (128x32 f32 -> bf16 LDS, packed b32 stores) ---
        #pragma unroll
        for (int p = 0; p < 4; ++p) {
            int row = p * 32 + (tid >> 3);
            int c4  = (tid & 7) * 4;
            const float* ga = A + (long)(m0 + row) * lda + k0 + c4;
            float a0 = ga[0], a1 = ga[1], a2 = ga[2], a3 = ga[3];
            unsigned* la = (unsigned*)&As[row * LDSS + c4];
            la[0] = pack2bf(a0, a1);
            la[1] = pack2bf(a2, a3);
        }
        // --- stage W tile (32x128 f32 -> bf16 LDS, transposed to [n][k]) ---
        // coalesced global loads (128 consecutive lanes per k-row), packed stores
        {
            const float* gw = W + (long)(k0 + bk) * ldw + n0 + bn;
            float wv16[16];
            #pragma unroll
            for (int j = 0; j < 16; ++j) wv16[j] = gw[(long)j * ldw];
            unsigned* lb = (unsigned*)&Bs[bn * LDSS + bk];
            #pragma unroll
            for (int j = 0; j < 8; ++j)
                lb[j] = pack2bf(wv16[2 * j], wv16[2 * j + 1]);
        }
        __syncthreads();

        Frag16 fa[2], fb[4];
        #pragma unroll
        for (int tm = 0; tm < 2; ++tm) {
            const unsigned short* pa = As + (wm + tm * 16 + lr) * LDSS + kh;
            fa[tm].q[0] = *(const uint4*)pa;
            fa[tm].q[1] = *(const uint4*)(pa + 16);
        }
        #pragma unroll
        for (int tn = 0; tn < 4; ++tn) {
            const unsigned short* pb = Bs + (wn + tn * 16 + lr) * LDSS + kh;
            fb[tn].q[0] = *(const uint4*)pb;
            fb[tn].q[1] = *(const uint4*)(pb + 16);
        }
        #pragma unroll
        for (int tm = 0; tm < 2; ++tm)
            #pragma unroll
            for (int tn = 0; tn < 4; ++tn)
                acc[tm][tn] = __builtin_amdgcn_wmma_f32_16x16x32_bf16(
                    false, fa[tm].v, false, fb[tn].v,
                    (short)0, acc[tm][tn], false, false);
        __syncthreads();
    }
}

// C[z] = act( A1*W1[z] (+ A2*W2[z]) + bias[z] ), M fixed by grid.y, N by grid.x
__global__ __launch_bounds__(256) void gemm_wmma_kernel(
    const float* __restrict__ A1, int lda1, const float* __restrict__ W1, int ldw1, long sW1,
    const float* __restrict__ A2, int lda2, const float* __restrict__ W2, int ldw2, long sW2,
    const float* __restrict__ bias, long sBias,
    float* __restrict__ C, int ldc, long sC,
    int K1, int K2, int act)
{
    __shared__ __align__(16) unsigned short As[TM * LDSS];
    __shared__ __align__(16) unsigned short Bs[TN * LDSS];

    const int m0 = blockIdx.y * TM;
    const int n0 = blockIdx.x * TN;
    const int z  = blockIdx.z;

    v8f acc[2][4];
    const v8f vzero = {0.f,0.f,0.f,0.f,0.f,0.f,0.f,0.f};
    #pragma unroll
    for (int tm = 0; tm < 2; ++tm)
        #pragma unroll
        for (int tn = 0; tn < 4; ++tn) acc[tm][tn] = vzero;

    gemm_ktiles(A1, lda1, W1 + (long)z * sW1, ldw1, K1, m0, n0, As, Bs, acc);
    if (A2)
        gemm_ktiles(A2, lda2, W2 + (long)z * sW2, ldw2, K2, m0, n0, As, Bs, acc);

    const int tid  = threadIdx.x;
    const int lane = tid & 31;
    const int wv   = tid >> 5;
    const int wm   = (wv & 3) * 32;
    const int wn   = (wv >> 2) * 64;
    const int lr   = lane & 15;
    const int rsel = (lane >> 4) * 8;   // C layout: VGPR i -> M=i (lanes 0-15), M=8+i (16-31)
    const float* bz = bias ? bias + (long)z * sBias : nullptr;
    float* Cz = C + (long)z * sC;

    #pragma unroll
    for (int tm = 0; tm < 2; ++tm)
        #pragma unroll
        for (int tn = 0; tn < 4; ++tn) {
            int coln = n0 + wn + tn * 16 + lr;
            float bv = bz ? bz[coln] : 0.f;
            #pragma unroll
            for (int i = 0; i < 8; ++i) {
                int row = m0 + wm + tm * 16 + rsel + i;
                float v = acc[tm][tn][i] + bv;
                if (act) v = (v > 0.f) ? v : 0.3f * v;   // leaky_relu(0.3)
                Cz[(long)row * ldc + coln] = v;
            }
        }
}

// ------------------------------- elementwise -------------------------------

static __device__ __forceinline__ float sigm(float x) { return 1.f / (1.f + expf(-x)); }

__global__ void select_sc_kernel(const float* __restrict__ S1, const float* __restrict__ S2,
                                 const float* __restrict__ g, float* __restrict__ shat, int n)
{
    int i = blockIdx.x * blockDim.x + threadIdx.x;
    if (i < n) shat[i] = (g[i] > 0.f) ? S1[i] : S2[i];
}

__global__ void combine_sv_kernel(const float* __restrict__ S1, const float* __restrict__ S2,
                                  const float* __restrict__ inputs, float* __restrict__ shat,
                                  int U_, int ldin, int n)
{
    int i = blockIdx.x * blockDim.x + threadIdx.x;
    if (i >= n) return;
    int b = i / U_;
    float click = inputs[(long)b * ldin + (ldin - 1)];
    shat[i] = S1[i] + ((click >= 0.5f) ? S2[i] : 0.f);
}

__global__ void cell_c_kernel(const float* __restrict__ PRE, long sPre,
                              const float* __restrict__ g,
                              const float* __restrict__ c_t_c, const float* __restrict__ c_c_hat,
                              float* __restrict__ cc_new, float* __restrict__ cchat_new,
                              float* __restrict__ ctc_new, float* __restrict__ hc_new, int n)
{
    int i = blockIdx.x * blockDim.x + threadIdx.x;
    if (i >= n) return;
    float ig = sigm(PRE[i]);
    float ih = sigm(PRE[sPre + i]);
    float f  = sigm(PRE[2 * sPre + i]);
    float fh = sigm(PRE[3 * sPre + i]);
    float o  = sigm(PRE[4 * sPre + i]);
    float z  = tanhf(PRE[5 * sPre + i]);
    float t  = 5.f * sigm(PRE[6 * sPre + i] * 0.2f);
    bool gm = g[i] > 0.f;
    float ccn  = ig * z + (gm ? 0.f : f * c_t_c[i]);
    float cchn = ih * z + (gm ? 0.f : fh * c_c_hat[i]);
    float ctn  = cchn + (ccn - cchn) * expf(-t);
    cc_new[i] = ccn; cchat_new[i] = cchn; ctc_new[i] = ctn;
    hc_new[i] = o * tanhf(ctn);
}

__global__ void cell_v_kernel(const float* __restrict__ PRE, long sPre,
                              const float* __restrict__ inputs,
                              const float* __restrict__ c_v, const float* __restrict__ c_v_hat,
                              const float* __restrict__ c_t_v, const float* __restrict__ h_v,
                              float* __restrict__ cv_new, float* __restrict__ cvhat_new,
                              float* __restrict__ ctv_new, float* __restrict__ hv_new,
                              float* __restrict__ g_new, int U_, int ldin, int n)
{
    int i = blockIdx.x * blockDim.x + threadIdx.x;
    if (i >= n) return;
    int b = i / U_;
    float click = inputs[(long)b * ldin + (ldin - 1)];
    bool gv = click >= 0.5f;
    float ig = sigm(PRE[i]);
    float ih = sigm(PRE[sPre + i]);
    float f  = sigm(PRE[2 * sPre + i]);
    float fh = sigm(PRE[3 * sPre + i]);
    float o  = sigm(PRE[4 * sPre + i]);
    float z  = tanhf(PRE[5 * sPre + i]);
    float t  = 5.f * sigm(PRE[6 * sPre + i] * 0.2f);
    float cvn  = gv ? (f * c_t_v[i] + ig * z)    : c_v[i];
    float cvhn = gv ? (fh * c_v_hat[i] + ih * z) : c_v_hat[i];
    float ctn  = cvhn + (cvn - cvhn) * expf(-t);
    float hvn  = gv ? (o * tanhf(ctn)) : h_v[i];
    cv_new[i] = cvn; cvhat_new[i] = cvhn; ctv_new[i] = ctn; hv_new[i] = hvn;
    g_new[i] = gv ? 1.f : 0.f;
}

// out[b] = sigmoid(dot(A[b,:K], W) + b0) (* mul[b] if mul)
__global__ void fc_kernel(const float* __restrict__ A, const float* __restrict__ W,
                          const float* __restrict__ b0, const float* __restrict__ mul,
                          float* __restrict__ out, int Kd, int n)
{
    int i = blockIdx.x * blockDim.x + threadIdx.x;
    if (i >= n) return;
    const float* row = A + (long)i * Kd;
    float s = 0.f;
    for (int j = 0; j < Kd; ++j) s += row[j] * W[j];
    s = 1.f / (1.f + expf(-(s + b0[0])));
    out[i] = mul ? s * mul[i] : s;
}

// --------------------------------- launch ----------------------------------

extern "C" void kernel_launch(void* const* d_in, const int* in_sizes, int n_in,
                              void* d_out, int out_size, void* d_ws, size_t ws_size,
                              hipStream_t stream)
{
    constexpr int B = 8192, D = 512, U = 512, P0 = 256, P1 = 128;
    constexpr int LDIN = D + 1;
    const long BU = (long)B * U;

    const float* inputs  = (const float*)d_in[0];
    const float* h_c     = (const float*)d_in[1];
    const float* h_v     = (const float*)d_in[2];
    const float* c_v     = (const float*)d_in[4];
    const float* c_c_hat = (const float*)d_in[5];
    const float* c_v_hat = (const float*)d_in[6];
    const float* c_t_c   = (const float*)d_in[7];
    const float* c_t_v   = (const float*)d_in[8];
    const float* g       = (const float*)d_in[9];
    const float* Wx_c    = (const float*)d_in[10];
    const float* bx_c    = (const float*)d_in[11];
    const float* Wh_c    = (const float*)d_in[12];
    const float* Wx_v    = (const float*)d_in[13];
    const float* bx_v    = (const float*)d_in[14];
    const float* Wh_v    = (const float*)d_in[15];
    const float* Ws_cc   = (const float*)d_in[16];
    const float* Ws_cv   = (const float*)d_in[17];
    const float* Ws_vv   = (const float*)d_in[18];
    const float* Ws_vc   = (const float*)d_in[19];
    const float* Pc_W0   = (const float*)d_in[20];
    const float* Pc_b0   = (const float*)d_in[21];
    const float* Pc_W1   = (const float*)d_in[22];
    const float* Pc_b1   = (const float*)d_in[23];
    const float* fc_c_W  = (const float*)d_in[24];
    const float* fc_c_b  = (const float*)d_in[25];
    const float* Pv_W0   = (const float*)d_in[26];
    const float* Pv_b0   = (const float*)d_in[27];
    const float* Pv_W1   = (const float*)d_in[28];
    const float* Pv_b1   = (const float*)d_in[29];
    const float* fc_v_W  = (const float*)d_in[30];
    const float* fc_v_b  = (const float*)d_in[31];

    float* out = (float*)d_out;
    float* ws  = (float*)d_ws;

    // workspace
    float* S1   = ws;            // B*U
    float* S2   = ws + BU;       // B*U
    float* SHAT = ws + 2 * BU;   // B*U
    float* PRE  = ws + 3 * BU;   // 7 * B*U
    float* A0   = S1;            // reuse for MLP (B*P0 <= B*U)
    float* A1   = S2;            // reuse for MLP (B*P1 <= B*U)

    // output layout (flat, return order)
    float* o_hcp   = out;
    float* o_hvp   = out + B;
    float* o_hc    = out + 2L * B;
    float* o_hv    = o_hc + BU;
    float* o_cc    = o_hc + 2 * BU;
    float* o_cv    = o_hc + 3 * BU;
    float* o_cchat = o_hc + 4 * BU;
    float* o_cvhat = o_hc + 5 * BU;
    float* o_ctc   = o_hc + 6 * BU;
    float* o_ctv   = o_hc + 7 * BU;
    float* o_g     = o_hc + 8 * BU;

    const dim3 blk(256);
    const int EW_GRID = (int)((BU + 255) / 256);

    auto gemm = [&](const float* A1p, int lda1, const float* W1p, int ldw1, long sW1,
                    const float* A2p, int lda2, const float* W2p, int ldw2, long sW2,
                    const float* bp, long sB, float* Cp, int ldc, long sC,
                    int N, int K1, int K2, int act, int nz) {
        dim3 gr(N / TN, B / TM, nz);
        gemm_wmma_kernel<<<gr, blk, 0, stream>>>(A1p, lda1, W1p, ldw1, sW1,
                                                 A2p, lda2, W2p, ldw2, sW2,
                                                 bp, sB, Cp, ldc, sC, K1, K2, act);
    };

    // ---- branch c ----
    gemm(h_v, U, Ws_cv, U, 0, nullptr, 0, nullptr, 0, 0, nullptr, 0, S1, U, 0, U, U, 0, 0, 1);
    gemm(h_c, U, Ws_cc, U, 0, nullptr, 0, nullptr, 0, 0, nullptr, 0, S2, U, 0, U, U, 0, 0, 1);
    select_sc_kernel<<<EW_GRID, blk, 0, stream>>>(S1, S2, g, SHAT, (int)BU);

    // all 7 gate pre-activations in one launch: x@Wx_c[k] + SHAT@Wh_c[k] + bx_c[k]
    gemm(inputs, LDIN, Wx_c, U, (long)D * U,
         SHAT, U, Wh_c, U, (long)U * U,
         bx_c, U, PRE, U, BU, U, D, U, 0, 7);

    cell_c_kernel<<<EW_GRID, blk, 0, stream>>>(PRE, BU, g, c_t_c, c_c_hat,
                                               o_cc, o_cchat, o_ctc, o_hc, (int)BU);

    // predict c
    gemm(o_hc, U, Pc_W0, P0, 0, nullptr, 0, nullptr, 0, 0, Pc_b0, 0, A0, P0, 0, P0, U, 0, 1, 1);
    gemm(A0, P0, Pc_W1, P1, 0, nullptr, 0, nullptr, 0, 0, Pc_b1, 0, A1, P1, 0, P1, P0, 0, 1, 1);
    fc_kernel<<<(B + 255) / 256, blk, 0, stream>>>(A1, fc_c_W, fc_c_b, nullptr, o_hcp, P1, B);

    // ---- branch v ----
    gemm(h_v, U, Ws_vv, U, 0, nullptr, 0, nullptr, 0, 0, nullptr, 0, S1, U, 0, U, U, 0, 0, 1);
    gemm(o_hc, U, Ws_vc, U, 0, nullptr, 0, nullptr, 0, 0, nullptr, 0, S2, U, 0, U, U, 0, 0, 1);
    combine_sv_kernel<<<EW_GRID, blk, 0, stream>>>(S1, S2, inputs, SHAT, U, LDIN, (int)BU);

    gemm(inputs, LDIN, Wx_v, U, (long)D * U,
         SHAT, U, Wh_v, U, (long)U * U,
         bx_v, U, PRE, U, BU, U, D, U, 0, 7);

    cell_v_kernel<<<EW_GRID, blk, 0, stream>>>(PRE, BU, inputs, c_v, c_v_hat, c_t_v, h_v,
                                               o_cv, o_cvhat, o_ctv, o_hv, o_g, U, LDIN, (int)BU);

    // predict v (multiplied by h_c_p)
    gemm(o_hv, U, Pv_W0, P0, 0, nullptr, 0, nullptr, 0, 0, Pv_b0, 0, A0, P0, 0, P0, U, 0, 1, 1);
    gemm(A0, P0, Pv_W1, P1, 0, nullptr, 0, nullptr, 0, 0, Pv_b1, 0, A1, P1, 0, P1, P0, 0, 1, 1);
    fc_kernel<<<(B + 255) / 256, blk, 0, stream>>>(A1, fc_v_W, fc_v_b, o_hcp, o_hvp, P1, B);

    (void)in_sizes; (void)n_in; (void)out_size; (void)ws_size;
}
